// GNN_23587960390209
// MI455X (gfx1250) — matrix operand order repaired and checked by reading, not compiled, not run on hardware
//
#include <hip/hip_runtime.h>

typedef __bf16 bf16;
typedef __attribute__((ext_vector_type(16))) __bf16 v16bf;
typedef __attribute__((ext_vector_type(8)))  float  v8f;

#define NNODES 100000
#define NEDGES 1250000
#define DIN    64
#define HDIM   128
#define EPSBN  1e-5f
#define SWS    272   // LDS B stride (bf16 elems): 256 + 16 pad, keeps 32B alignment, avoids bank conflicts

// ---------------- utility kernels ----------------

__global__ void k_zero(float* p, int n) {
  int i = blockIdx.x * blockDim.x + threadIdx.x;
  int stride = gridDim.x * blockDim.x;
  for (; i < n; i += stride) p[i] = 0.f;
}

__global__ void k_deg(const int* __restrict__ dst, float* deg, int e) {
  int i = blockIdx.x * blockDim.x + threadIdx.x;
  if (i < e) atomicAdd(&deg[dst[i]], 1.0f);
}

__global__ void k_invdeg(float* deg, int n) {
  int i = blockIdx.x * blockDim.x + threadIdx.x;
  if (i < n) deg[i] = 1.0f / fmaxf(deg[i], 1.0f);
}

// agg[dst] += x[src]  (float4 per thread, 4 global f32 atomics)
__global__ void k_scatter(const float* __restrict__ x, const int* __restrict__ src,
                          const int* __restrict__ dst, float* agg,
                          int k4shift, int K, int total) {
  int i = blockIdx.x * blockDim.x + threadIdx.x;
  if (i >= total) return;
  int e = i >> k4shift;
  int c = i & ((1 << k4shift) - 1);
  int s = src[e], d = dst[e];
  const float4 v = ((const float4*)(x + (size_t)s * K))[c];
  float* a = agg + (size_t)d * K + c * 4;
  atomicAdd(a + 0, v.x); atomicAdd(a + 1, v.y);
  atomicAdd(a + 2, v.z); atomicAdd(a + 3, v.w);
}

// pack combined weight [W_self ; W_neigh] (2K x 128) to bf16
__global__ void k_prepw(const float* __restrict__ Ws, const float* __restrict__ Wn,
                        bf16* Wc, int K, int total) {
  int i = blockIdx.x * blockDim.x + threadIdx.x;
  if (i >= total) return;
  int k = i >> 7;          // / 128
  int n = i & 127;
  float v = (k < K) ? Ws[k * HDIM + n] : Wn[(k - K) * HDIM + n];
  Wc[i] = (bf16)v;
}

// ---------------- WMMA GEMM: h = [x | agg*invdeg] @ Wcomb + bias, fused BN partial sums --------
// block = 256 threads (8 waves). Block tile: 128 rows x 64 cols.
// wave w -> rows [blockIdx.x*128 + w*16, +16), 4 col-tiles of 16 within 64-col slab blockIdx.y.
// Templated on K so the k-loop fully unrolls (compile-time region split, pipelined B ds_loads).
template <int K>
__global__ __launch_bounds__(256) void k_gemm(
    const float* __restrict__ actIn, const float* __restrict__ agg,
    const float* __restrict__ invdeg, const bf16* __restrict__ Wc,
    const float* __restrict__ bias, float* __restrict__ h,
    float* colSum, float* colSq, int nNodes)
{
  __shared__ __align__(32) bf16 sW[64 * SWS];   // transposed: sW[col*SWS + k]
  __shared__ float sSum[64], sSq[64];
  constexpr int K2 = K * 2;
  const int tid  = threadIdx.x;
  const int wave = tid >> 5, lane = tid & 31;
  const int rowBase = blockIdx.x * 128 + wave * 16;
  const int colBase = blockIdx.y * 64;

  // stage weight slab to LDS, transposed (coalesced global reads)
  #pragma unroll
  for (int idx0 = 0; idx0 < K2 * 64; idx0 += 256) {
    int idx = idx0 + tid;
    int k = idx >> 6, c = idx & 63;
    sW[c * SWS + k] = Wc[k * HDIM + colBase + c];
  }
  if (tid < 64) { sSum[tid] = 0.f; sSq[tid] = 0.f; }
  __syncthreads();

  // A-fragment lane mapping (16-bit A 16x32): m = lane%16, element j -> k = (lane/16)*16 + j
  const int m = lane & 15, half = lane >> 4;
  const int row  = rowBase + m;
  const int rowc = (row < nNodes) ? row : (nNodes - 1);
  const float idg = invdeg[rowc];
  const float* xrow = actIn + (size_t)rowc * K;
  const float* grow = agg   + (size_t)rowc * K;

  const v8f vzero = {0.f,0.f,0.f,0.f,0.f,0.f,0.f,0.f};
  v8f acc[4];
  #pragma unroll
  for (int t = 0; t < 4; ++t) acc[t] = vzero;

  #pragma unroll
  for (int ks = 0; ks < (K2 >> 5); ++ks) {
    constexpr int unused = 0; (void)unused;
    const int kb = ks * 32;                        // compile-time after unroll
    const bool isN = (kb >= K);                    // compile-time region split
    const float* srcp = isN ? (grow + (kb - K)) : (xrow + kb);
    float4 f0 = *(const float4*)(srcp + half * 16);
    float4 f1 = *(const float4*)(srcp + half * 16 + 4);
    float4 f2 = *(const float4*)(srcp + half * 16 + 8);
    float4 f3 = *(const float4*)(srcp + half * 16 + 12);
    v16bf a;
    if (isN) {
      a[0]=(bf16)(f0.x*idg); a[1]=(bf16)(f0.y*idg); a[2]=(bf16)(f0.z*idg); a[3]=(bf16)(f0.w*idg);
      a[4]=(bf16)(f1.x*idg); a[5]=(bf16)(f1.y*idg); a[6]=(bf16)(f1.z*idg); a[7]=(bf16)(f1.w*idg);
      a[8]=(bf16)(f2.x*idg); a[9]=(bf16)(f2.y*idg); a[10]=(bf16)(f2.z*idg); a[11]=(bf16)(f2.w*idg);
      a[12]=(bf16)(f3.x*idg); a[13]=(bf16)(f3.y*idg); a[14]=(bf16)(f3.z*idg); a[15]=(bf16)(f3.w*idg);
    } else {
      a[0]=(bf16)f0.x; a[1]=(bf16)f0.y; a[2]=(bf16)f0.z; a[3]=(bf16)f0.w;
      a[4]=(bf16)f1.x; a[5]=(bf16)f1.y; a[6]=(bf16)f1.z; a[7]=(bf16)f1.w;
      a[8]=(bf16)f2.x; a[9]=(bf16)f2.y; a[10]=(bf16)f2.z; a[11]=(bf16)f2.w;
      a[12]=(bf16)f3.x; a[13]=(bf16)f3.y; a[14]=(bf16)f3.z; a[15]=(bf16)f3.w;
    }
    #pragma unroll
    for (int ct = 0; ct < 4; ++ct) {
      // B lane mapping (16-bit B 32x16): n = lane%16, element j -> k = (lane/16)*16 + j
      const int lc = ct * 16 + m;
      const v16bf b = *(const v16bf*)&sW[lc * SWS + kb + half * 16];
      acc[ct] = __builtin_amdgcn_wmma_f32_16x16x32_bf16(
          false, a, false, b, (short)0, acc[ct], false, false);
    }
  }

  // preload bias for the 4 col-tiles (overlaps with trailing WMMAs)
  float bv[4];
  #pragma unroll
  for (int ct = 0; ct < 4; ++ct) bv[ct] = bias[colBase + ct * 16 + m];

  // epilogue: C lane mapping (f32 16x16): vgpr r, lane -> row=(lane/16)*8+r, col=lane%16
  const int rEdge = nNodes - (rowBase + half * 8);   // #valid rows in this lane's half-tile
  if (rEdge >= 8) {
    // fast path: full tile, wave-uniform branch, no per-element predication
    #pragma unroll
    for (int ct = 0; ct < 4; ++ct) {
      const int lc   = ct * 16 + m;
      const int colG = colBase + lc;
      float s1 = 0.f, s2 = 0.f;
      #pragma unroll
      for (int r = 0; r < 8; ++r) {
        float v = acc[ct][r] + bv[ct];
        h[(size_t)(rowBase + half * 8 + r) * HDIM + colG] = v;
        s1 += v; s2 += v * v;
      }
      atomicAdd(&sSum[lc], s1);
      atomicAdd(&sSq[lc], s2);
    }
  } else {
    // tail block only (1 of 782 row-blocks)
    #pragma unroll
    for (int ct = 0; ct < 4; ++ct) {
      const int lc   = ct * 16 + m;
      const int colG = colBase + lc;
      float s1 = 0.f, s2 = 0.f;
      for (int r = 0; r < 8; ++r) {
        if (r < rEdge) {
          float v = acc[ct][r] + bv[ct];
          h[(size_t)(rowBase + half * 8 + r) * HDIM + colG] = v;
          s1 += v; s2 += v * v;
        }
      }
      atomicAdd(&sSum[lc], s1);
      atomicAdd(&sSq[lc], s2);
    }
  }
  __syncthreads();
  if (tid < 64) {
    atomicAdd(&colSum[colBase + tid], sSum[tid]);
    atomicAdd(&colSq[colBase + tid],  sSq[tid]);
  }
}

// BN: scale = gamma*rsqrt(var+eps), shift = beta - mu*scale
__global__ void k_bnfin(const float* colSum, const float* colSq,
                        const float* __restrict__ gamma, const float* __restrict__ beta,
                        float* ss, float invN) {
  int n = threadIdx.x;
  if (n < HDIM) {
    float mu  = colSum[n] * invN;
    float var = fmaxf(colSq[n] * invN - mu * mu, 0.f);
    float s   = gamma[n] * rsqrtf(var + EPSBN);
    ss[n]        = s;
    ss[HDIM + n] = beta[n] - mu * s;
  }
}

// in-place h = relu(h*scale + shift)
__global__ void k_apply(float* h, const float* __restrict__ ss, int total4) {
  int i = blockIdx.x * blockDim.x + threadIdx.x;
  if (i >= total4) return;
  float4 v = ((float4*)h)[i];
  int c = (i * 4) & (HDIM - 1);
  v.x = fmaxf(v.x * ss[c + 0] + ss[HDIM + c + 0], 0.f);
  v.y = fmaxf(v.y * ss[c + 1] + ss[HDIM + c + 1], 0.f);
  v.z = fmaxf(v.z * ss[c + 2] + ss[HDIM + c + 2], 0.f);
  v.w = fmaxf(v.w * ss[c + 3] + ss[HDIM + c + 3], 0.f);
  ((float4*)h)[i] = v;
}

// out[n, 0:2] = act[n,:] @ Wc + bc
__global__ void k_cls(const float* __restrict__ act, const float* __restrict__ Wc,
                      const float* __restrict__ bc, float* out, int n) {
  int i = blockIdx.x * blockDim.x + threadIdx.x;
  if (i >= n) return;
  const float4* a = (const float4*)(act + (size_t)i * HDIM);
  float a0 = bc[0], a1 = bc[1];
  #pragma unroll 8
  for (int j4 = 0; j4 < HDIM / 4; ++j4) {
    float4 v = a[j4];
    int j = j4 * 4;
    a0 += v.x * Wc[2*j+0] + v.y * Wc[2*j+2] + v.z * Wc[2*j+4] + v.w * Wc[2*j+6];
    a1 += v.x * Wc[2*j+1] + v.y * Wc[2*j+3] + v.z * Wc[2*j+5] + v.w * Wc[2*j+7];
  }
  out[2 * i]     = a0;
  out[2 * i + 1] = a1;
}

// ---------------- launcher ----------------

extern "C" void kernel_launch(void* const* d_in, const int* in_sizes, int n_in,
                              void* d_out, int out_size, void* d_ws, size_t ws_size,
                              hipStream_t stream) {
  const int N = NNODES, E = NEDGES;
  const float* x   = (const float*)d_in[0];
  const int*   src = (const int*)d_in[1];
  const int*   dst = (const int*)d_in[2];

  char* wsb = (char*)d_ws;
  size_t off = 0;
  auto take = [&](size_t bytes) -> char* {
    char* p = wsb + off;
    off = (off + bytes + 255) & ~(size_t)255;
    return p;
  };
  float* AGG = (float*)take((size_t)N * HDIM * 4);
  float* P   = (float*)take((size_t)N * HDIM * 4);
  float* Q   = (float*)take((size_t)N * HDIM * 4);
  float* INV = (float*)take((size_t)N * 4);
  float* CS  = (float*)take(256 * 4);     // colSum[128] | colSq[128]
  float* SS  = (float*)take(256 * 4);     // scale[128] | shift[128]
  bf16*  WC  = (bf16*)take((size_t)256 * HDIM * 2);

  const int T = 256;

  // degree -> invdeg (graph is identical across layers)
  k_zero  <<<1024, T, 0, stream>>>(INV, N);
  k_deg   <<<(E + T - 1) / T, T, 0, stream>>>(dst, INV, E);
  k_invdeg<<<(N + T - 1) / T, T, 0, stream>>>(INV, N);

  for (int L = 0; L < 3; ++L) {
    const int K = (L == 0) ? DIN : HDIM;
    const float* actIn = (L == 0) ? x : ((L == 1) ? P : Q);
    float*       hOut  = (L == 0) ? P : ((L == 1) ? Q : P);
    const float* Ws = (const float*)d_in[3 + 5 * L];
    const float* bs = (const float*)d_in[4 + 5 * L];
    const float* Wn = (const float*)d_in[5 + 5 * L];
    const float* ga = (const float*)d_in[6 + 5 * L];
    const float* be = (const float*)d_in[7 + 5 * L];

    k_zero<<<2048, T, 0, stream>>>(AGG, N * K);
    const int k4s = (K == 64) ? 4 : 5;
    const int totalS = E * (K / 4);
    k_scatter<<<(totalS + T - 1) / T, T, 0, stream>>>(actIn, src, dst, AGG, k4s, K, totalS);

    k_zero<<<1, T, 0, stream>>>(CS, 256);
    const int totW = 2 * K * HDIM;
    k_prepw<<<(totW + T - 1) / T, T, 0, stream>>>(Ws, Wn, WC, K, totW);

    dim3 g((N + 127) / 128, 2);
    if (K == DIN) {
      k_gemm<DIN><<<g, T, 0, stream>>>(actIn, AGG, INV, WC, bs, hOut, CS, CS + HDIM, N);
    } else {
      k_gemm<HDIM><<<g, T, 0, stream>>>(actIn, AGG, INV, WC, bs, hOut, CS, CS + HDIM, N);
    }

    k_bnfin<<<1, HDIM, 0, stream>>>(CS, CS + HDIM, ga, be, SS, 1.0f / (float)N);
    const int total4 = N * HDIM / 4;
    k_apply<<<(total4 + T - 1) / T, T, 0, stream>>>(hOut, SS, total4);
  }

  const float* Wcls = (const float*)d_in[18];
  const float* bcls = (const float*)d_in[19];
  k_cls<<<(N + T - 1) / T, T, 0, stream>>>(P, Wcls, bcls, (float*)d_out, N);
}
